// ExportGatedDeltaNet_8100308320385
// MI455X (gfx1250) — matrix operand order repaired and checked by reading, not compiled, not run on hardware
//
#include <hip/hip_runtime.h>

// ---------------------------------------------------------------------------
// GatedDeltaNet forward for MI455X (gfx1250, wave32, WMMA + TDM).
// bf16 matrix math (v_wmma_f32_16x16x32_bf16, f32 accumulate), f32 state in
// LDS, Tensor Data Mover staging of strided Q/K/V chunk tiles into LDS.
// ---------------------------------------------------------------------------

typedef __bf16 bf16;
typedef bf16  v16bf __attribute__((ext_vector_type(16)));
typedef float v8f   __attribute__((ext_vector_type(8)));
typedef unsigned int v4u __attribute__((ext_vector_type(4)));
typedef unsigned int v8u __attribute__((ext_vector_type(8)));

constexpr int NKH = 16, NVH = 32, DKD = 128, DVD = 128, KC = 4, CDIM = 2048;
constexpr int KEY_DIM  = NKH * DKD;            // 2048
constexpr int VAL_DIM  = NVH * DVD;            // 4096
constexpr int CONV_DIM = 2 * KEY_DIM + VAL_DIM;// 8192
constexpr int BB = 2, TT = 2048;
constexpr int MTOT = BB * TT;                  // 4096
constexpr int CHUNK = 64;
constexpr int NCHUNK = TT / CHUNK;             // 32
constexpr float EPSF = 1e-6f;

// ---------------------------------------------------------------------------
// WMMA helpers (16x16x32 bf16 -> f32). A/B fragments use contiguous k = 16*hf + i.
// ---------------------------------------------------------------------------
__device__ __forceinline__ v8f wmma_bf16(v16bf a, v16bf b, v8f c) {
  return __builtin_amdgcn_wmma_f32_16x16x32_bf16(false, a, false, b,
                                                 (short)0, c, false, false);
}

__device__ __forceinline__ v16bf load_a_frag(const bf16* buf, int stride,
                                             int row, int k0, int hf) {
  v16bf a;
#pragma unroll
  for (int i = 0; i < 16; ++i) {
    int kk = 16 * hf + i;
    a[i] = buf[row * stride + k0 + kk];
  }
  return a;
}

// B operand stored as [N][K] (row = output column)
__device__ __forceinline__ v16bf load_b_frag_nk(const bf16* buf, int stride,
                                                int col, int k0, int hf) {
  v16bf b;
#pragma unroll
  for (int i = 0; i < 16; ++i) {
    int kk = 16 * hf + i;
    b[i] = buf[col * stride + k0 + kk];
  }
  return b;
}

// B operand stored as [K][N]
__device__ __forceinline__ v16bf load_b_frag_kn(const bf16* buf, int stride,
                                                int k0, int col, int hf) {
  v16bf b;
#pragma unroll
  for (int i = 0; i < 16; ++i) {
    int kk = 16 * hf + i;
    b[i] = buf[(k0 + kk) * stride + col];
  }
  return b;
}

// ---------------------------------------------------------------------------
// Tensor Data Mover: 2D tile (tile_w bf16 per row x tile_h rows) from global
// memory (row stride = row_stride_elems bf16) into LDS at lds_byte_off, with
// LDS padding 4 DWORDs every 64 DWORDs (row 256B -> 272B, i.e. 136-elem rows).
// D# layout per CDNA5 ISA. Issued per-wave; caller must
// make the call wave-uniform and follow with s_wait_tensorcnt + barrier.
// ---------------------------------------------------------------------------
__device__ __forceinline__ void tdm_load_tile_bf16(const void* gptr,
                                                   unsigned lds_byte_off,
                                                   unsigned row_stride_elems,
                                                   unsigned tile_w,
                                                   unsigned tile_h) {
  unsigned long long ga = (unsigned long long)gptr;
  v4u g0;
  g0.x = 1u;                                        // count=1 user descriptor
  g0.y = lds_byte_off;                              // lds_addr [63:32]
  g0.z = (unsigned)ga;                              // global_addr [95:64]
  g0.w = (unsigned)((ga >> 32) & 0x01FFFFFFu)       // global_addr[56:32]
         | (2u << 30);                              // type=2 ("image")
  v8u g1;
  g1.s0 = (1u << 16)                                // data_size=1 (2 bytes)
        | (1u << 20)                                // pad_enable
        | (5u << 22)                                // pad_interval: 64 DWORDs
        | (3u << 25);                               // pad_amount: 4 DWORDs
  g1.s1 = 0u;                                       // abar addr, tdim0[15:0]=0
  g1.s2 = 0x4000u;                                  // tensor_dim0 = 1<<30
  g1.s3 = 0x4000u | (tile_w << 16);                 // tensor_dim1=1<<30, tile0
  g1.s4 = tile_h;                                   // tile_dim1, tile_dim2=0
  g1.s5 = row_stride_elems;                         // tensor_dim0_stride lo
  g1.s6 = 0u;                                       // stride hi, dim1_stride lo
  g1.s7 = 0u;                                       // dim1_stride hi
  asm volatile("tensor_load_to_lds %0, %1" ::"s"(g0), "s"(g1) : "memory");
}

// ---------------------------------------------------------------------------
// Generic GEMM: out[M,N] (f32) = A[M,K] x W[N,K]^T, A is f32 or bf16, W f32.
// 128x128 block tile, K-step 32, 8 waves in a 4(M) x 2(N) grid, each wave
// computes a 32x64 patch = 2x4 WMMA tiles.
// ---------------------------------------------------------------------------
constexpr int BM = 128, BN = 128, BK = 32;
constexpr int LSTR = 40;  // bf16 LDS row stride (pad vs. bank conflicts)

template <typename TA>
__global__ __launch_bounds__(256) void gemm_xWt_kernel(
    const TA* __restrict__ A, const float* __restrict__ W,
    float* __restrict__ out, int M, int N, int Kd) {
  __shared__ bf16 As[BM * LSTR];
  __shared__ bf16 Bs[BN * LSTR];
  const int tid  = threadIdx.x;
  const int lane = tid & 31, w = tid >> 5;
  const int hf = (lane >> 4) & 1, ln = lane & 15;
  const int m0 = blockIdx.y * BM, n0 = blockIdx.x * BN;
  const int wm = w & 3, wn = w >> 2;

  v8f acc[2][4];
  for (int i = 0; i < 2; ++i)
    for (int j = 0; j < 4; ++j) acc[i][j] = {};

  const int lrow = tid >> 1;
  const int lcol = (tid & 1) * 16;

  for (int k0 = 0; k0 < Kd; k0 += BK) {
    __syncthreads();
    {
      const TA* ap = A + (size_t)(m0 + lrow) * Kd + k0 + lcol;
      const float* wp = W + (size_t)(n0 + lrow) * Kd + k0 + lcol;
#pragma unroll
      for (int u = 0; u < 16; ++u) As[lrow * LSTR + lcol + u] = (bf16)(float)ap[u];
#pragma unroll
      for (int u = 0; u < 16; ++u) Bs[lrow * LSTR + lcol + u] = (bf16)wp[u];
      if (k0 + BK < Kd) {  // gfx1250 global_prefetch_b8 for next tiles
        __builtin_prefetch(ap + BK, 0, 0);
        __builtin_prefetch(wp + BK, 0, 0);
      }
    }
    __syncthreads();

    v16bf af[2], bfr[4];
    for (int mt = 0; mt < 2; ++mt)
      af[mt] = load_a_frag(As, LSTR, wm * 32 + mt * 16 + ln, 0, hf);
    for (int nt = 0; nt < 4; ++nt)
      bfr[nt] = load_b_frag_nk(Bs, LSTR, wn * 64 + nt * 16 + ln, 0, hf);
    for (int mt = 0; mt < 2; ++mt)
      for (int nt = 0; nt < 4; ++nt)
        acc[mt][nt] = wmma_bf16(af[mt], bfr[nt], acc[mt][nt]);
  }

  for (int mt = 0; mt < 2; ++mt)
    for (int nt = 0; nt < 4; ++nt)
#pragma unroll
      for (int j = 0; j < 8; ++j) {
        int row = m0 + wm * 32 + mt * 16 + 8 * hf + j;
        int col = n0 + wn * 64 + nt * 16 + ln;
        out[(size_t)row * N + col] = acc[mt][nt][j];
      }
}

// ---------------------------------------------------------------------------
// beta = sigmoid(x W_b^T); g = -exp(A_log) * softplus(x W_a^T + dt_bias)
// ---------------------------------------------------------------------------
__global__ __launch_bounds__(256) void proj_beta_g_kernel(
    const float* __restrict__ x, const float* __restrict__ Wb,
    const float* __restrict__ Wa, const float* __restrict__ dt_bias,
    const float* __restrict__ A_log, float* __restrict__ beta_out,
    float* __restrict__ g_out) {
  __shared__ float xrow[CDIM];
  __shared__ float part[64 * 4];
  const int m = blockIdx.x;
  const int tid = threadIdx.x;
  for (int i = tid; i < CDIM; i += 256) xrow[i] = x[(size_t)m * CDIM + i];
  __syncthreads();
  const int h64 = tid & 63, p = tid >> 6;
  const float* wr = (h64 < NVH) ? (Wb + (size_t)h64 * CDIM)
                                : (Wa + (size_t)(h64 - NVH) * CDIM);
  float s = 0.f;
  const int kseg = CDIM / 4;
  for (int k = p * kseg; k < (p + 1) * kseg; ++k) s += xrow[k] * wr[k];
  part[h64 * 4 + p] = s;
  __syncthreads();
  if (tid < 64) {
    float t = part[tid * 4] + part[tid * 4 + 1] + part[tid * 4 + 2] +
              part[tid * 4 + 3];
    if (tid < NVH) {
      beta_out[(size_t)m * NVH + tid] = 1.f / (1.f + __expf(-t));
    } else {
      int h = tid - NVH;
      float av = t + dt_bias[h];
      float sp = (av > 20.f) ? av : log1pf(__expf(av));
      g_out[(size_t)m * NVH + h] = -__expf(A_log[h]) * sp;
    }
  }
}

// ---------------------------------------------------------------------------
// Depthwise causal conv (K=4) + SiLU, then per-head L2-norm of q,k.
// ---------------------------------------------------------------------------
__global__ __launch_bounds__(256) void conv_silu_norm_kernel(
    const float* __restrict__ qkv, const float* __restrict__ conv_w,
    const float* __restrict__ conv_state, const int* __restrict__ input_pos,
    bf16* __restrict__ qn, bf16* __restrict__ kn, bf16* __restrict__ vn) {
  __shared__ float sbuf[CONV_DIM];
  const int m = blockIdx.x;  // b*T + t
  const int b = m / TT, t = m % TT;
  const int tid = threadIdx.x;
  const float keep = (input_pos[0] == 0) ? 0.f : 1.f;

  for (int d = tid; d < CONV_DIM; d += 256) {
    float accv = 0.f;
#pragma unroll
    for (int j = 0; j < KC; ++j) {
      int ci = t + 1 + j;  // index into [conv_state | qkv] timeline
      float tap;
      if (ci < KC)
        tap = keep * conv_state[((size_t)b * CONV_DIM + d) * KC + ci];
      else
        tap = qkv[((size_t)b * TT + (ci - KC)) * CONV_DIM + d];
      accv += conv_w[d * KC + j] * tap;
    }
    sbuf[d] = accv / (1.f + __expf(-accv));  // SiLU
  }
  __syncthreads();

  const int w = tid >> 5, lane = tid & 31;
  for (int i = 0; i < 4; ++i) {
    int head = w * 4 + i;  // 0..31: 16 q heads then 16 k heads
    const float* src = (head < NKH) ? (sbuf + head * DKD)
                                    : (sbuf + KEY_DIM + (head - NKH) * DKD);
    float vals[4];
    float ss = 0.f;
#pragma unroll
    for (int jj = 0; jj < 4; ++jj) {
      vals[jj] = src[lane + 32 * jj];
      ss += vals[jj] * vals[jj];
    }
#pragma unroll
    for (int off = 16; off > 0; off >>= 1) ss += __shfl_xor(ss, off, 32);
    float scale = 1.f / fmaxf(sqrtf(ss), 1e-12f);
    bf16* dst = (head < NKH)
                    ? (qn + ((size_t)m * NKH + head) * DKD)
                    : (kn + ((size_t)m * NKH + (head - NKH)) * DKD);
#pragma unroll
    for (int jj = 0; jj < 4; ++jj) dst[lane + 32 * jj] = (bf16)(vals[jj] * scale);
  }
  for (int i = tid; i < VAL_DIM; i += 256)
    vn[(size_t)m * VAL_DIM + i] = (bf16)sbuf[2 * KEY_DIM + i];
}

// ---------------------------------------------------------------------------
// Chunked gated-linear-attention recurrence, one block per (b, head).
// State S[128][128] f32 in LDS; TDM stages Q/K/V chunk tiles; four WMMA
// stages per 64-step chunk.
// ---------------------------------------------------------------------------
constexpr int SSTR = 132;  // f32 row stride for S
constexpr int QSTR = 136;  // bf16 row stride for Q/K/V tiles (TDM pad: 64+4 DW)
constexpr int PSTR = 72;   // bf16 row stride for P

constexpr size_t ATTN_LDS = (size_t)DKD * SSTR * 4 + (size_t)3 * CHUNK * QSTR * 2 +
                            (size_t)CHUNK * PSTR * 2 + 4 * CHUNK * 4;

__global__ __launch_bounds__(256) void attn_chunk_kernel(
    const bf16* __restrict__ qn, const bf16* __restrict__ kn,
    const bf16* __restrict__ vn, const float* __restrict__ beta,
    const float* __restrict__ g, const float* __restrict__ rs0,
    const int* __restrict__ input_pos, float* __restrict__ o) {
  extern __shared__ char smem[];
  float* S    = (float*)smem;                             // 128 x SSTR f32
  bf16*  Qc   = (bf16*)(smem + (size_t)DKD * SSTR * 4);   // 64 x QSTR
  bf16*  Kc   = Qc + CHUNK * QSTR;
  bf16*  Vc   = Kc + CHUNK * QSTR;
  bf16*  P    = Vc + CHUNK * QSTR;                        // 64 x PSTR
  float* Gc   = (float*)(P + CHUNK * PSTR);               // 64
  float* expg = Gc + CHUNK;
  float* coef = expg + CHUNK;
  float* betal = coef + CHUNK;

  const int bh = blockIdx.x;
  const int b = bh / NVH, h = bh % NVH, hq = h >> 1;  // rep = NV/NK = 2
  const int tid = threadIdx.x;
  const int w = tid >> 5, lane = tid & 31;
  const int hf = lane >> 4, ln = lane & 15;
  const float keep = (input_pos[0] == 0) ? 0.f : 1.f;

  // LDS byte offsets for TDM destinations (flat->LDS: low 32 bits = offset)
  const unsigned ldsQ = (unsigned)(unsigned long long)(void*)Qc;
  const unsigned ldsK = (unsigned)(unsigned long long)(void*)Kc;
  const unsigned ldsV = (unsigned)(unsigned long long)(void*)Vc;

  for (int i = tid; i < DKD * DVD; i += 256) {
    int dk = i >> 7, dv = i & 127;
    S[dk * SSTR + dv] = keep * rs0[(((size_t)b * NVH + h) * DKD + dk) * DVD + dv];
  }
  __syncthreads();

  for (int c = 0; c < NCHUNK; ++c) {
    const int t0 = c * CHUNK;
    // ---- stage Q/K/V chunk tiles into LDS with the Tensor Data Mover ----
    if (w == 0) {  // wave-uniform: single wave issues the three DMAs
      const bf16* qg = qn + ((size_t)(b * TT + t0) * NKH + hq) * DKD;
      const bf16* kg = kn + ((size_t)(b * TT + t0) * NKH + hq) * DKD;
      const bf16* vg = vn + ((size_t)(b * TT + t0) * NVH + h) * DVD;
      tdm_load_tile_bf16(qg, ldsQ, NKH * DKD, DKD, CHUNK);
      tdm_load_tile_bf16(kg, ldsK, NKH * DKD, DKD, CHUNK);
      tdm_load_tile_bf16(vg, ldsV, NVH * DVD, DVD, CHUNK);
      __builtin_amdgcn_s_wait_tensorcnt(0);
    }
    if (tid < CHUNK)
      betal[tid] = beta[(size_t)(b * TT + t0 + tid) * NVH + h];
    if (tid == 0) {
      float run = 0.f;
      for (int s = 0; s < CHUNK; ++s) {
        run += g[(size_t)(b * TT + t0 + s) * NVH + h];
        Gc[s] = run;  // inclusive prefix, all <= 0
      }
    }
    __syncthreads();
    if (tid < CHUNK) {
      expg[tid] = __expf(Gc[tid]);
      coef[tid] = betal[tid] * __expf(Gc[CHUNK - 1] - Gc[tid]);
    }
    __syncthreads();

    // ---- P = (Q K^T) with causal decay mask * beta  (16 tiles, 2/wave) ----
    for (int q = 0; q < 2; ++q) {
      int tile = w * 2 + q;
      int tr0 = (tile >> 2) * 16, tc0 = (tile & 3) * 16;
      v8f acc = {};
      for (int kt = 0; kt < 4; ++kt) {
        v16bf a  = load_a_frag(Qc, QSTR, tr0 + ln, kt * 32, hf);
        v16bf bb = load_b_frag_nk(Kc, QSTR, tc0 + ln, kt * 32, hf);
        acc = wmma_bf16(a, bb, acc);
      }
#pragma unroll
      for (int j = 0; j < 8; ++j) {
        int tr = tr0 + 8 * hf + j, tc = tc0 + ln;
        float val = (tc <= tr)
                        ? acc[j] * __expf(Gc[tr] - Gc[tc]) * betal[tc]
                        : 0.f;
        P[tr * PSTR + tc] = (bf16)val;
      }
    }
    __syncthreads();

    // ---- out = P V + (Q * e^G) S_in  (32 tiles, 4/wave) ----
    for (int q = 0; q < 4; ++q) {
      int tile = w * 4 + q;
      int tr0 = (tile >> 3) * 16, tc0 = (tile & 7) * 16;
      v8f acc = {};
      for (int kt = 0; kt < 2; ++kt) {  // intra-chunk term
        v16bf a  = load_a_frag(P, PSTR, tr0 + ln, kt * 32, hf);
        v16bf bb = load_b_frag_kn(Vc, QSTR, kt * 32, tc0 + ln, hf);
        acc = wmma_bf16(a, bb, acc);
      }
      for (int kt = 0; kt < 4; ++kt) {  // carried-state term
        int row = tr0 + ln;
        float eg = expg[row];
        v16bf a, bb;
#pragma unroll
        for (int i = 0; i < 16; ++i) {
          int kk = 16 * hf + i;
          a[i] = (bf16)((float)Qc[row * QSTR + kt * 32 + kk] * eg);
          int kb = kt * 32 + 16 * hf + i;
          bb[i] = (bf16)S[kb * SSTR + tc0 + ln];
        }
        acc = wmma_bf16(a, bb, acc);
      }
#pragma unroll
      for (int j = 0; j < 8; ++j) {
        int tr = tr0 + 8 * hf + j, tc = tc0 + ln;
        o[(((size_t)(b * TT + t0 + tr)) * NVH + h) * DVD + tc] = acc[j];
      }
    }
    __syncthreads();

    // ---- S = e^{G_L} S + K_hat^T V  (64 tiles, 8/wave) ----
    float decay = __expf(Gc[CHUNK - 1]);
    for (int q = 0; q < 8; ++q) {
      int tile = w * 8 + q;
      int tr0 = (tile >> 3) * 16, tc0 = (tile & 7) * 16;
      v8f acc = {};
      for (int kt = 0; kt < 2; ++kt) {
        v16bf a, bb;
#pragma unroll
        for (int i = 0; i < 16; ++i) {
          int ks = kt * 32 + 16 * hf + i;
          a[i] = (bf16)((float)Kc[ks * QSTR + tr0 + ln] * coef[ks]);
          int kb = kt * 32 + 16 * hf + i;
          bb[i] = Vc[kb * QSTR + tc0 + ln];
        }
        acc = wmma_bf16(a, bb, acc);
      }
#pragma unroll
      for (int j = 0; j < 8; ++j) {
        int mr = tr0 + 8 * hf + j, nc = tc0 + ln;
        S[mr * SSTR + nc] = S[mr * SSTR + nc] * decay + acc[j];
      }
    }
    __syncthreads();
  }
}

// ---------------------------------------------------------------------------
// Gated RMSNorm: o *= silu(z); o *= rsqrt(mean(o^2)+eps) * norm_w  -> bf16
// ---------------------------------------------------------------------------
__global__ __launch_bounds__(128) void gated_rmsnorm_kernel(
    const float* __restrict__ o, const float* __restrict__ z,
    const float* __restrict__ norm_w, bf16* __restrict__ on) {
  __shared__ float red[4];
  const int mh = blockIdx.x;  // m*NVH + h
  const int d = threadIdx.x;
  const size_t idx = (size_t)mh * DVD + d;
  float zz = z[idx];
  float val = o[idx] * (zz / (1.f + __expf(-zz)));
  float ss = val * val;
#pragma unroll
  for (int off = 16; off > 0; off >>= 1) ss += __shfl_xor(ss, off, 32);
  if ((threadIdx.x & 31) == 0) red[threadIdx.x >> 5] = ss;
  __syncthreads();
  float var = (red[0] + red[1] + red[2] + red[3]) * (1.f / DVD);
  on[idx] = (bf16)(val * rsqrtf(var + EPSF) * norm_w[d]);
}

// ---------------------------------------------------------------------------
// Host-side orchestration
// ---------------------------------------------------------------------------
extern "C" void kernel_launch(void* const* d_in, const int* in_sizes, int n_in,
                              void* d_out, int out_size, void* d_ws,
                              size_t ws_size, hipStream_t stream) {
  (void)in_sizes; (void)n_in; (void)out_size; (void)ws_size;
  const float* x          = (const float*)d_in[0];
  const int*   input_pos  = (const int*)d_in[1];
  const float* W_qkv      = (const float*)d_in[2];
  const float* W_z        = (const float*)d_in[3];
  const float* W_b        = (const float*)d_in[4];
  const float* W_a        = (const float*)d_in[5];
  const float* conv_w     = (const float*)d_in[6];
  const float* dt_bias    = (const float*)d_in[7];
  const float* A_log      = (const float*)d_in[8];
  const float* norm_w     = (const float*)d_in[9];
  const float* W_out      = (const float*)d_in[10];
  const float* conv_state = (const float*)d_in[11];
  const float* rec_state  = (const float*)d_in[12];
  float* out = (float*)d_out;

  // Workspace layout (bytes). qkv region (128MB) is reused for o (64MB) and
  // on (32MB) once the conv stage has consumed qkv.
  char* ws = (char*)d_ws;
  constexpr size_t SZ_QKV  = (size_t)MTOT * CONV_DIM * 4;  // 128MB
  constexpr size_t SZ_O    = (size_t)MTOT * VAL_DIM * 4;   // 64MB (alias @0)
  constexpr size_t OFF_Z    = SZ_QKV;
  constexpr size_t SZ_Z     = (size_t)MTOT * VAL_DIM * 4;
  constexpr size_t OFF_BETA = OFF_Z + SZ_Z;
  constexpr size_t SZ_BETA  = (size_t)MTOT * NVH * 4;
  constexpr size_t OFF_G    = OFF_BETA + SZ_BETA;
  constexpr size_t OFF_QN   = OFF_G + SZ_BETA;
  constexpr size_t SZ_QN    = (size_t)MTOT * NKH * DKD * 2;
  constexpr size_t OFF_KN   = OFF_QN + SZ_QN;
  constexpr size_t OFF_VN   = OFF_KN + SZ_QN;

  float* qkv  = (float*)ws;
  float* ob   = (float*)ws;                 // alias over dead qkv
  bf16*  on   = (bf16*)(ws + SZ_O);         // alias, after o region
  float* zb   = (float*)(ws + OFF_Z);
  float* betab = (float*)(ws + OFF_BETA);
  float* gb   = (float*)(ws + OFF_G);
  bf16*  qnb  = (bf16*)(ws + OFF_QN);
  bf16*  knb  = (bf16*)(ws + OFF_KN);
  bf16*  vnb  = (bf16*)(ws + OFF_VN);

  // 1) qkv = x @ W_qkv^T   [4096 x 8192 x 2048]
  gemm_xWt_kernel<float><<<dim3(CONV_DIM / BN, MTOT / BM), 256, 0, stream>>>(
      x, W_qkv, qkv, MTOT, CONV_DIM, CDIM);
  // 2) z = x @ W_z^T       [4096 x 4096 x 2048]
  gemm_xWt_kernel<float><<<dim3(VAL_DIM / BN, MTOT / BM), 256, 0, stream>>>(
      x, W_z, zb, MTOT, VAL_DIM, CDIM);
  // 3) beta, g
  proj_beta_g_kernel<<<MTOT, 256, 0, stream>>>(x, W_b, W_a, dt_bias, A_log,
                                               betab, gb);
  // 4) conv + silu + q/k L2 norm -> bf16 q,k,v
  conv_silu_norm_kernel<<<MTOT, 256, 0, stream>>>(qkv, conv_w, conv_state,
                                                  input_pos, qnb, knb, vnb);
  // 5) chunked gated-linear-attention recurrence (TDM-staged tiles)
  attn_chunk_kernel<<<BB * NVH, 256, ATTN_LDS, stream>>>(
      qnb, knb, vnb, betab, gb, rec_state, input_pos, ob);
  // 6) gated RMSNorm -> bf16
  gated_rmsnorm_kernel<<<MTOT * NVH, 128, 0, stream>>>(ob, zb, norm_w, on);
  // 7) out = on @ W_out^T  [4096 x 2048 x 4096]
  gemm_xWt_kernel<bf16><<<dim3(CDIM / BN, MTOT / BM), 256, 0, stream>>>(
      on, W_out, out, MTOT, CDIM, VAL_DIM);
}
